// RigidDocking_4183298146991
// MI455X (gfx1250) — matrix operand order repaired and checked by reading, not compiled, not run on hardware
//
#include <hip/hip_runtime.h>
#include <hip/hip_bf16.h>

// ---------------------------------------------------------------------------
// SAKE rigid-docking forward, fused for MI455X (gfx1250, wave32, WMMA).
// Heavy per-edge dense layers run on v_wmma_f32_16x16x32_f16 with f32 accum.
// Each workgroup owns one node i and sweeps neighbors j in tiles of 16, so
// every (n,n,*) intermediate stays in LDS.
//
// Round-2: WMMA weights pre-swizzled into the ISA 32x16 B-fragment lane
// layout (one aligned 32B read per lane per fragment) + K-tile prefetch.
// Round-3: hot-loop transcendentals use HW TRANS ops (__expf/__cosf/v_rcp)
// instead of IEEE division chains, so they co-execute with the WMMA pipe.
// ---------------------------------------------------------------------------

#define N0V   384
#define N1V   128
#define HIDN  64
#define NCN   256
#define RBFN  50
#define CUTF  5.0f
#define PI_F  3.14159265358979323846f
#define PIOC  0.62831853071795864769f   // pi / CUT

typedef __attribute__((ext_vector_type(16))) _Float16 v16h;
typedef __attribute__((ext_vector_type(8)))  float    v8f;

__device__ __forceinline__ float rcp_f(float v)   { return __builtin_amdgcn_rcpf(v); }
__device__ __forceinline__ float silu_f(float v)  { return v * rcp_f(1.f + __expf(-v)); }
__device__ __forceinline__ float celu2_f(float v) { return v > 0.f ? v : 2.f * (__expf(0.5f * v) - 1.f); }

__device__ __forceinline__ v8f zero8() {
    v8f z = {0.f, 0.f, 0.f, 0.f, 0.f, 0.f, 0.f, 0.f};
    return z;
}

// A fragment: 16x32 f16 (MxK), row-major source with leading dim lda.
// Per CDNA5 ISA 7.12.2: elements 0..7 = A[m][8*hf + 0..7] (contiguous),
// elements 8..15 = A[m][16 + 8*hf + 0..7] (contiguous) -> two b128 LDS loads.
__device__ __forceinline__ v16h frag_a(const _Float16* A, int lda) {
    int lane = threadIdx.x & 31;
    int m = lane & 15, hf = lane >> 4;
    union { v16h v; _Float16 h[16]; } a;
#pragma unroll
    for (int r = 0; r < 8; ++r) {
        int kb = (r < 4) ? (2 * r + 8 * hf) : (16 + 2 * (r - 4) + 8 * hf);
        a.h[2 * r]     = A[m * lda + kb];
        a.h[2 * r + 1] = A[m * lda + kb + 1];
    }
    return a.v;
}

// Packed-B GEMM: D(16x16) += A(16xK) * Wp[:, ntile], K multiple of 32.
// Wp layout: tile t = kt*ntiles + ntile holds 32 lanes x 16 halves,
// lane-contiguous, exactly the ISA 32x16 B-fragment register image.
__device__ __forceinline__ v8f gemm16p(const _Float16* A, int lda,
                                       const _Float16* Wp, int ntiles,
                                       int ntile, int K, v8f acc) {
    const int lane = threadIdx.x & 31;
    const int nkt = K >> 5;
    for (int kt = 0; kt < nkt; ++kt) {
        const _Float16* wp =
            Wp + (((size_t)kt * ntiles + ntile) * 32 + lane) * 16;
        if (kt + 1 < nkt)
            __builtin_prefetch(wp + (size_t)ntiles * 512, 0, 1);  // next K-tile
        v16h a = frag_a(A + kt * 32, lda);
        v16h b = *(const v16h*)wp;  // 32B aligned, coalesced
        acc = __builtin_amdgcn_wmma_f32_16x16x32_f16(false, a, false, b,
                                                     (short)0, acc, false, false);
    }
    return acc;
}

// ---------------------------------------------------------------------------
// Fused SAKE layer: one workgroup per node i, 128 threads (4 wave32).
// ---------------------------------------------------------------------------
struct SakeArgs {
    const float*    h;      // n x 64 current features (f32)
    const _Float16* hF16;   // n x 64 (f16 copy)
    const float*    x;      // n x 3
    int             n;
    const _Float16* Wcfin;  // packed, logical 128 x 64 (cols 50..63 zero)
    const float*    bcfin;  // 50
    const float*    means;  // 50
    const float*    betas;  // 50
    const _Float16* Wo1;    // packed, logical 192 x 64 (rows 179..191 zero)
    const float*    bo1;    // 64
    const _Float16* Wo2;    // packed, logical 64 x 64
    const float*    bo2;    // 64
    const float*    semW;   // 64 x 4
    const float*    semb;   // 4
    const _Float16* Wxmix;  // packed, logical 256 x 256
    const float*    p1W;    // 256 x 64
    const float*    p1b;
    const float*    p2W;    // 64 x 64
    const float*    p2b;
    const float*    n1W;    // 384 x 64
    const float*    n1b;
    const float*    n2W;    // 64 x 64
    const float*    n2b;
    float*          hOut;   // n x 64
};

__global__ __launch_bounds__(128) void sake_layer_kernel(SakeArgs A) {
    extern __shared__ char smem_raw[];
    const int n    = A.n;
    const int i    = blockIdx.x;
    const int tid  = threadIdx.x;
    const int wave = tid >> 5;
    const int lane = tid & 31;
    const int m16  = lane & 15;
    const int hf   = lane >> 4;

    char* pp = smem_raw;
    auto take = [&](size_t b) -> char* {
        char* r = pp;
        pp += (b + 15) & ~(size_t)15;
        return r;
    };
    _Float16* heAll  = (_Float16*)take((size_t)2 * 64 * n);   // h_e, f16
    float*    dAll   = (float*)take((size_t)4 * n);
    float*    dirAll = (float*)take((size_t)12 * n);
    float*    slAll  = (float*)take((size_t)16 * n);          // celu(sem logits)
    float*    attAll = (float*)take((size_t)16 * n);          // final attention
    _Float16* Atile  = (_Float16*)take(16 * 192 * 2);         // edge activations
    _Float16* eTile  = (_Float16*)take(16 * 64 * 2);
    _Float16* heaT   = (_Float16*)take(16 * 256 * 2);         // h_e_att tile
    float*    comb   = (float*)take(256 * 3 * 4);
    float*    hagg   = (float*)take(256 * 4);
    float*    scr    = (float*)take(1024 * 4);
    float*    red    = (float*)take(64 * 4);

    const float xi0 = A.x[3 * i + 0];
    const float xi1 = A.x[3 * i + 1];
    const float xi2 = A.x[3 * i + 2];
    const int nt = n >> 4;

    // zero accumulators
    for (int c = tid; c < 256; c += 128) {
        hagg[c] = 0.f;
        comb[3 * c + 0] = 0.f;
        comb[3 * c + 1] = 0.f;
        comb[3 * c + 2] = 0.f;
    }

    // ------------------------- Pass A: edge MLPs --------------------------
    for (int jt = 0; jt < nt; ++jt) {
        const int j0 = jt << 4;
        // h_cat = [h_i | h_j] into Atile cols 0..127 (lda = 192)
        for (int idx = tid; idx < 16 * 128; idx += 128) {
            int m = idx >> 7, k = idx & 127;
            Atile[m * 192 + k] = (k < 64) ? A.hF16[(size_t)i * 64 + k]
                                          : A.hF16[(size_t)(j0 + m) * 64 + (k - 64)];
        }
        if (tid < 16) {
            int j = j0 + tid;
            float dx = A.x[3 * j + 0] - xi0;
            float dy = A.x[3 * j + 1] - xi1;
            float dz = A.x[3 * j + 2] - xi2;
            float dd = sqrtf(dx * dx + dy * dy + dz * dz + 1e-5f);
            dAll[j] = dd;
            float inv = rcp_f(dd + 1e-5f);
            dirAll[3 * j + 0] = dx * inv;
            dirAll[3 * j + 1] = dy * inv;
            dirAll[3 * j + 2] = dz * inv;
        }
        __syncthreads();

        // cf_in: 16x128 @ 128x64 (cols >=50 zero-padded); ntiles=4, per-wave tile
        v8f acc = zero8();
        acc = gemm16p(Atile, 192, A.Wcfin, 4, wave, 128, acc);
        {
            int c = wave * 16 + m16;
            union { v8f v; float f[8]; } cf; cf.v = acc;
#pragma unroll
            for (int r = 0; r < 8; ++r) {
                int m = r + 8 * hf;
                float v = 0.f;
                if (c < RBFN) {
                    float hin = cf.f[r] + A.bcfin[c];
                    float dd  = dAll[j0 + m];
                    float cut = (dd < CUTF) ? 0.5f * (__cosf(dd * PIOC) + 1.f) : 0.f;
                    float t   = __expf(-dd) - A.means[c];
                    v = cut * __expf(-A.betas[c] * t * t) * hin;
                }
                Atile[m * 192 + 128 + c] = (_Float16)v;  // cols 128..191
            }
        }
        __syncthreads();
        if (tid < 16) Atile[tid * 192 + 178] = (_Float16)dAll[j0 + tid];  // d column
        __syncthreads();

        // cf_o1: 16x192 @ 192x64, +bias, silu
        v8f a2 = zero8();
        a2 = gemm16p(Atile, 192, A.Wo1, 4, wave, 192, a2);
        {
            int c = wave * 16 + m16;
            union { v8f v; float f[8]; } cf; cf.v = a2;
#pragma unroll
            for (int r = 0; r < 8; ++r) {
                int m = r + 8 * hf;
                eTile[m * 64 + c] = (_Float16)silu_f(cf.f[r] + A.bo1[c]);
            }
        }
        __syncthreads();

        // cf_o2: 16x64 @ 64x64, +bias -> h_e
        v8f a3 = zero8();
        a3 = gemm16p(eTile, 64, A.Wo2, 4, wave, 64, a3);
        {
            int c = wave * 16 + m16;
            union { v8f v; float f[8]; } cf; cf.v = a3;
#pragma unroll
            for (int r = 0; r < 8; ++r) {
                int m = r + 8 * hf;
                heAll[(size_t)(j0 + m) * 64 + c] = (_Float16)(cf.f[r] + A.bo2[c]);
            }
        }
        __syncthreads();

        // sem logits + celu(alpha=2)
        if (tid < 64) {
            int m = tid >> 2, hd = tid & 3;
            const _Float16* he = &heAll[(size_t)(j0 + m) * 64];
            float s = A.semb[hd];
            for (int k = 0; k < 64; ++k) s += (float)he[k] * A.semW[k * 4 + hd];
            slAll[(j0 + m) * 4 + hd] = celu2_f(s);
        }
        __syncthreads();
    }

    // ------------------ Pass B: softmaxes over j (fixed i) ----------------
    if (tid < 5) {
        int ch = tid;  // 0 = euclidean, 1..4 = heads
        float mx = -1e30f;
        for (int j = 0; j < n; ++j) {
            float l = (ch == 0) ? -dAll[j] : slAll[j * 4 + (ch - 1)];
            if (j == i) l -= 1e5f;
            mx = fmaxf(mx, l);
        }
        float s = 0.f;
        for (int j = 0; j < n; ++j) {
            float l = (ch == 0) ? -dAll[j] : slAll[j * 4 + (ch - 1)];
            if (j == i) l -= 1e5f;
            s += __expf(l - mx);
        }
        red[2 * ch] = mx;
        red[2 * ch + 1] = rcp_f(s);  // store reciprocal denominator
    }
    __syncthreads();
    for (int idx = tid; idx < n * 4; idx += 128) {
        int j = idx >> 2, hd = idx & 3;
        float le = -dAll[j] - ((j == i) ? 1e5f : 0.f);
        float pe = __expf(le - red[0]) * red[1];
        float ls = slAll[j * 4 + hd] - ((j == i) ? 1e5f : 0.f);
        float ps = __expf(ls - red[2 * (hd + 1)]) * red[2 * (hd + 1) + 1];
        attAll[idx] = pe * ps;
    }
    __syncthreads();
    if (tid < 4) {
        float s = 0.f;
        for (int j = 0; j < n; ++j) s += attAll[j * 4 + tid];
        red[16 + tid] = rcp_f(s);
    }
    __syncthreads();
    for (int idx = tid; idx < n * 4; idx += 128) attAll[idx] *= red[16 + (idx & 3)];
    __syncthreads();

    // ---------------- Pass C: xmix GEMM + spatial reduction ---------------
    for (int jt = 0; jt < nt; ++jt) {
        const int j0 = jt << 4;
        for (int idx = tid; idx < 16 * 256; idx += 128) {
            int m = idx >> 8, c = idx & 255;
            int j = j0 + m;
            float v = (float)heAll[(size_t)j * 64 + (c >> 2)] * attAll[j * 4 + (c & 3)];
            heaT[m * 256 + c] = (_Float16)v;
            atomicAdd(&hagg[c], v);  // h_agg = sum_j h_e_att
        }
        __syncthreads();
#pragma unroll
        for (int t4 = 0; t4 < 4; ++t4) {
            int n0 = (wave * 4 + t4) * 16;
            v8f acc = zero8();
            acc = gemm16p(heaT, 256, A.Wxmix, 16, wave * 4 + t4, 256, acc);
            union { v8f v; float f[8]; } cf; cf.v = acc;
#pragma unroll
            for (int r = 0; r < 8; ++r) {
                int m = r + 8 * hf;
                int c = n0 + m16;
                float co = tanhf(cf.f[r]);
                int j = j0 + m;
                atomicAdd(&comb[c * 3 + 0], co * dirAll[3 * j + 0]);
                atomicAdd(&comb[c * 3 + 1], co * dirAll[3 * j + 1]);
                atomicAdd(&comb[c * 3 + 2], co * dirAll[3 * j + 2]);
            }
        }
        __syncthreads();
    }

    // ---------------------- Pass D: node update MLPs ----------------------
    const float invn = rcp_f((float)n);
    for (int c = tid; c < 256; c += 128) {
        float a0 = comb[3 * c + 0] * invn;
        float a1 = comb[3 * c + 1] * invn;
        float a2 = comb[3 * c + 2] * invn;
        scr[c] = a0 * a0 + a1 * a1 + a2 * a2;
    }
    __syncthreads();
    if (tid < 64) {  // post1
        float s = A.p1b[tid];
        for (int k = 0; k < 256; ++k) s += scr[k] * A.p1W[k * 64 + tid];
        scr[256 + tid] = silu_f(s);
    }
    __syncthreads();
    if (tid < 64) {  // post2 -> h_comb
        float s = A.p2b[tid];
        for (int k = 0; k < 64; ++k) s += scr[256 + k] * A.p2W[k * 64 + tid];
        scr[320 + tid] = silu_f(s);
    }
    __syncthreads();
    if (tid < 64) {  // node1 on [h | h_agg | h_comb]
        float s = A.n1b[tid];
        const float* hi = &A.h[(size_t)i * 64];
        for (int k = 0; k < 64;  ++k) s += hi[k]        * A.n1W[k * 64 + tid];
        for (int k = 0; k < 256; ++k) s += hagg[k]      * A.n1W[(64 + k) * 64 + tid];
        for (int k = 0; k < 64;  ++k) s += scr[320 + k] * A.n1W[(320 + k) * 64 + tid];
        scr[384 + tid] = silu_f(s);
    }
    __syncthreads();
    if (tid < 64) {  // node2 + residual
        float s = A.n2b[tid];
        for (int k = 0; k < 64; ++k) s += scr[384 + k] * A.n2W[k * 64 + tid];
        A.hOut[(size_t)i * 64 + tid] = A.h[(size_t)i * 64 + tid] + silu_f(s);
    }
}

// ---------------------------------------------------------------------------
// Small helper kernels
// ---------------------------------------------------------------------------
__global__ void dense_kernel(const float* in, const float* W, const float* b,
                             float* out, int n, int K, int O, int act) {
    int idx = blockIdx.x * blockDim.x + threadIdx.x;
    if (idx >= n * O) return;
    int i = idx / O, c = idx % O;
    float s = b ? b[c] : 0.f;
    const float* row = in + (size_t)i * K;
    for (int k = 0; k < K; ++k) s += row[k] * W[k * O + c];
    if (act) s = silu_f(s);
    out[idx] = s;
}

__global__ void f32_to_f16_kernel(const float* src, _Float16* dst, int count) {
    int idx = blockIdx.x * blockDim.x + threadIdx.x;
    if (idx < count) dst[idx] = (_Float16)src[idx];
}

// Convert f32 weight (K x N) to f16 packed into the CDNA5 32x16 B-fragment
// lane image, zero-padded to (Kp x Np). dst index:
//   (((kt*ntiles + nt)*32 + lane)*16 + e)  with
//   k = kt*32 + (e&7) + 8*(lane>>4) + 16*(e>>3),  c = nt*16 + (lane&15)
__global__ void pack_b_kernel(const float* src, _Float16* dst,
                              int K, int N, int Kp, int Np) {
    int idx = blockIdx.x * blockDim.x + threadIdx.x;
    if (idx >= Kp * Np) return;
    int e    = idx & 15;
    int lane = (idx >> 4) & 31;
    int t    = idx >> 9;
    int ntiles = Np >> 4;
    int kt = t / ntiles, nt = t % ntiles;
    int k = kt * 32 + (e & 7) + 8 * (lane >> 4) + 16 * (e >> 3);
    int c = nt * 16 + (lane & 15);
    dst[idx] = (k < K && c < N) ? (_Float16)src[k * N + c] : (_Float16)0.f;
}

__global__ void dot_kernel(const float* h0, const float* h1, float* dot,
                           int n0, int n1) {
    int idx = blockIdx.x * blockDim.x + threadIdx.x;
    if (idx >= n0 * n1) return;
    int i = idx / n1, j = idx % n1;
    float s = 0.f;
    for (int k = 0; k < 64; ++k) s += h0[(size_t)i * 64 + k] * h1[(size_t)j * 64 + k];
    dot[idx] = s * 0.125f;  // 1/sqrt(HID)
}

// cross message for molecule 0: softmax over N1 per i
__global__ __launch_bounds__(128) void msg0_kernel(const float* h0, const float* h1,
                                                   const float* dot, const float* Wc,
                                                   const float* bc, float* out,
                                                   int n0, int n1) {
    __shared__ float pa[512];
    __shared__ float cat[128];
    __shared__ float st[2];
    int i = blockIdx.x, tid = threadIdx.x;
    if (tid == 0) {
        float mx = -1e30f;
        for (int j = 0; j < n1; ++j) mx = fmaxf(mx, dot[(size_t)i * n1 + j]);
        float s = 0.f;
        for (int j = 0; j < n1; ++j) s += __expf(dot[(size_t)i * n1 + j] - mx);
        st[0] = mx; st[1] = __builtin_amdgcn_rcpf(s);
    }
    __syncthreads();
    for (int j = tid; j < n1; j += blockDim.x)
        pa[j] = __expf(dot[(size_t)i * n1 + j] - st[0]) * st[1];
    __syncthreads();
    if (tid < 64) {
        float s = 0.f;
        for (int j = 0; j < n1; ++j) s += pa[j] * h1[(size_t)j * 64 + tid];
        cat[tid]      = h0[(size_t)i * 64 + tid];
        cat[64 + tid] = h0[(size_t)i * 64 + tid] - s;  // m0
    }
    __syncthreads();
    if (tid < 64) {
        float s = bc[tid];
        for (int k = 0; k < 128; ++k) s += cat[k] * Wc[k * 64 + tid];
        out[(size_t)i * 64 + tid] = silu_f(s);
    }
}

// cross message for molecule 1: softmax over N0 per j
__global__ __launch_bounds__(128) void msg1_kernel(const float* h0, const float* h1,
                                                   const float* dot, const float* Wc,
                                                   const float* bc, float* out,
                                                   int n0, int n1) {
    __shared__ float pa[512];
    __shared__ float cat[128];
    __shared__ float st[2];
    int j = blockIdx.x, tid = threadIdx.x;
    if (tid == 0) {
        float mx = -1e30f;
        for (int i = 0; i < n0; ++i) mx = fmaxf(mx, dot[(size_t)i * n1 + j]);
        float s = 0.f;
        for (int i = 0; i < n0; ++i) s += __expf(dot[(size_t)i * n1 + j] - mx);
        st[0] = mx; st[1] = __builtin_amdgcn_rcpf(s);
    }
    __syncthreads();
    for (int i = tid; i < n0; i += blockDim.x)
        pa[i] = __expf(dot[(size_t)i * n1 + j] - st[0]) * st[1];
    __syncthreads();
    if (tid < 64) {
        float s = 0.f;
        for (int i = 0; i < n0; ++i) s += pa[i] * h0[(size_t)i * 64 + tid];
        cat[tid]      = h1[(size_t)j * 64 + tid];
        cat[64 + tid] = h1[(size_t)j * 64 + tid] - s;  // m1
    }
    __syncthreads();
    if (tid < 64) {
        float s = bc[tid];
        for (int k = 0; k < 128; ++k) s += cat[k] * Wc[k * 64 + tid];
        out[(size_t)j * 64 + tid] = silu_f(s);
    }
}

__global__ void transform_prep_kernel(const float* h, const float* x,
                                      const float* Wo, const float* Wx, const float* Wy,
                                      float* tOut, float* POut, int n) {
    int i = blockIdx.x * blockDim.x + threadIdx.x;
    if (i >= n) return;
    const float* hi = h + (size_t)i * 64;
    float so = 0.f, sx = 0.f, sy = 0.f;
    for (int k = 0; k < 64; ++k) {
        so += hi[k] * Wo[k];
        sx += hi[k] * Wx[k];
        sy += hi[k] * Wy[k];
    }
    float X[3], org[3], ax[3], ay[3], az[3];
    for (int a = 0; a < 3; ++a) X[a] = x[3 * i + a];
    for (int a = 0; a < 3; ++a) org[a] = so * X[a];
    for (int a = 0; a < 3; ++a) ax[a] = sx * X[a] - org[a];
    for (int a = 0; a < 3; ++a) ay[a] = sy * X[a] - org[a];
    float nx = 1.f / sqrtf(ax[0] * ax[0] + ax[1] * ax[1] + ax[2] * ax[2]);
    float ny = 1.f / sqrtf(ay[0] * ay[0] + ay[1] * ay[1] + ay[2] * ay[2]);
    for (int a = 0; a < 3; ++a) { ax[a] *= nx; ay[a] *= ny; }
    az[0] = ax[1] * ay[2] - ax[2] * ay[1];
    az[1] = ax[2] * ay[0] - ax[0] * ay[2];
    az[2] = ax[0] * ay[1] - ax[1] * ay[0];
    float M[3][3] = {{ax[0], ax[1], ax[2]}, {ay[0], ay[1], ay[2]}, {az[0], az[1], az[2]}};
    float c00 = M[1][1] * M[2][2] - M[1][2] * M[2][1];
    float c01 = M[1][0] * M[2][2] - M[1][2] * M[2][0];
    float c02 = M[1][0] * M[2][1] - M[1][1] * M[2][0];
    float det = M[0][0] * c00 - M[0][1] * c01 + M[0][2] * c02;
    float id = 1.f / det;
    float* P = POut + (size_t)i * 9;
    P[0] =  c00 * id;
    P[1] = (M[0][2] * M[2][1] - M[0][1] * M[2][2]) * id;
    P[2] = (M[0][1] * M[1][2] - M[0][2] * M[1][1]) * id;
    P[3] = -c01 * id;
    P[4] = (M[0][0] * M[2][2] - M[0][2] * M[2][0]) * id;
    P[5] = (M[0][2] * M[1][0] - M[0][0] * M[1][2]) * id;
    P[6] =  c02 * id;
    P[7] = (M[0][1] * M[2][0] - M[0][0] * M[2][1]) * id;
    P[8] = (M[0][0] * M[1][1] - M[0][1] * M[1][0]) * id;
    for (int a = 0; a < 3; ++a) tOut[3 * i + a] = X[a] - org[a];
}

// out[b,i,:] = t[i,:] @ P[b]   (faithful (n,3)@(n,3,3) -> (n,n,3) broadcast)
__global__ void transform_write_kernel(const float* t, const float* P,
                                       float* out, int n) {
    int idx = blockIdx.x * blockDim.x + threadIdx.x;
    if (idx >= n * n) return;
    int b = idx / n, i = idx % n;
    const float* Pb = P + (size_t)b * 9;
    const float* ti = t + (size_t)i * 3;
    for (int c = 0; c < 3; ++c)
        out[(size_t)idx * 3 + c] = ti[0] * Pb[c] + ti[1] * Pb[3 + c] + ti[2] * Pb[6 + c];
}

// ---------------------------------------------------------------------------
// Host orchestration
// ---------------------------------------------------------------------------
static size_t sake_shmem(int n) {
    auto a16 = [](size_t b) { return (b + 15) & ~(size_t)15; };
    size_t o = 0;
    o += a16((size_t)2 * 64 * n);   // heAll
    o += a16((size_t)4 * n);        // dAll
    o += a16((size_t)12 * n);       // dirAll
    o += a16((size_t)16 * n);       // slAll
    o += a16((size_t)16 * n);       // attAll
    o += a16(16 * 192 * 2);         // Atile
    o += a16(16 * 64 * 2);          // eTile
    o += a16(16 * 256 * 2);         // heaT
    o += a16(256 * 3 * 4);          // comb
    o += a16(256 * 4);              // hagg
    o += a16(1024 * 4);             // scr
    o += a16(64 * 4);               // red
    return o;
}

// Assumed flattening: setup_inputs() insertion order, depth-first.
#define P_EMB_IN_W 4
#define P_EMB_IN_B 5
#define P_EMB_O1_W 6
#define P_EMB_O1_B 7
#define P_EMB_O2_W 8
#define P_EMB_O2_B 9
#define P_LAYER(l) (10 + 19 * (l))
// layer leaf offsets: 0 cf_in.W, 1 cf_in.b, 2 means, 3 betas, 4 cf_o1.W, 5 cf_o1.b,
// 6 cf_o2.W, 7 cf_o2.b, 8 sem.W, 9 sem.b, 10 xmix.W, 11 post1.W, 12 post1.b,
// 13 post2.W, 14 post2.b, 15 node1.W, 16 node1.b, 17 node2.W, 18 node2.b
#define P_COMBINE_W 86
#define P_COMBINE_B 87
#define P_ORIG_W    88
#define P_XAX_W     89
#define P_YAX_W     90

extern "C" void kernel_launch(void* const* d_in, const int* in_sizes, int n_in,
                              void* d_out, int out_size, void* d_ws, size_t ws_size,
                              hipStream_t stream) {
    (void)in_sizes; (void)n_in; (void)out_size; (void)ws_size;

    const float* h0in = (const float*)d_in[0];
    const float* x0   = (const float*)d_in[1];
    const float* h1in = (const float*)d_in[2];
    const float* x1   = (const float*)d_in[3];
    auto PF = [&](int idx) { return (const float*)d_in[idx]; };

    // ---------------- workspace carve ----------------
    char*  base = (char*)d_ws;
    size_t off  = 0;
    auto alloc = [&](size_t bytes) -> void* {
        off = (off + 255) & ~(size_t)255;
        void* p = base + off;
        off += bytes;
        return p;
    };
    float* hA0  = (float*)alloc((size_t)N0V * 64 * 4);
    float* hB0  = (float*)alloc((size_t)N0V * 64 * 4);
    float* hA1  = (float*)alloc((size_t)N1V * 64 * 4);
    float* hB1  = (float*)alloc((size_t)N1V * 64 * 4);
    float* tmp0 = (float*)alloc((size_t)N0V * 64 * 4);
    float* tmp1 = (float*)alloc((size_t)N1V * 64 * 4);
    float* dotb = (float*)alloc((size_t)N0V * N1V * 4);
    float* t0   = (float*)alloc((size_t)N0V * 3 * 4);
    float* Pm0  = (float*)alloc((size_t)N0V * 9 * 4);
    float* t1   = (float*)alloc((size_t)N1V * 3 * 4);
    float* Pm1  = (float*)alloc((size_t)N1V * 9 * 4);
    _Float16* hF0 = (_Float16*)alloc((size_t)N0V * 64 * 2);
    _Float16* hF1 = (_Float16*)alloc((size_t)N1V * 64 * 2);
    _Float16* Wcfin[4], *Wo1[4], *Wo2[4], *Wxm[4];
    for (int l = 0; l < 4; ++l) {
        Wcfin[l] = (_Float16*)alloc(128 * 64 * 2);
        Wo1[l]   = (_Float16*)alloc(192 * 64 * 2);
        Wo2[l]   = (_Float16*)alloc(64 * 64 * 2);
        Wxm[l]   = (_Float16*)alloc(256 * 256 * 2);
    }

    // ------------- weight conversion: f32 -> packed f16 B-fragments -------
    for (int l = 0; l < 4; ++l) {
        int B = P_LAYER(l);
        pack_b_kernel<<<(128 * 64 + 255) / 256, 256, 0, stream>>>(
            PF(B + 0), Wcfin[l], 128, 50, 128, 64);
        pack_b_kernel<<<(192 * 64 + 255) / 256, 256, 0, stream>>>(
            PF(B + 4), Wo1[l], 179, 64, 192, 64);
        pack_b_kernel<<<(64 * 64 + 255) / 256, 256, 0, stream>>>(
            PF(B + 6), Wo2[l], 64, 64, 64, 64);
        pack_b_kernel<<<(256 * 256 + 255) / 256, 256, 0, stream>>>(
            PF(B + 10), Wxm[l], 256, 256, 256, 256);
    }

    // ---------------- input embedding ----------------
    dense_kernel<<<(N0V * 64 + 255) / 256, 256, 0, stream>>>(
        h0in, PF(P_EMB_IN_W), PF(P_EMB_IN_B), hA0, N0V, 32, 64, 0);
    dense_kernel<<<(N1V * 64 + 255) / 256, 256, 0, stream>>>(
        h1in, PF(P_EMB_IN_W), PF(P_EMB_IN_B), hA1, N1V, 32, 64, 0);

    float* cur0 = hA0; float* nxt0 = hB0;
    float* cur1 = hA1; float* nxt1 = hB1;
    const size_t sh0 = sake_shmem(N0V);
    const size_t sh1 = sake_shmem(N1V);

    for (int l = 0; l < 4; ++l) {
        int B = P_LAYER(l);
        f32_to_f16_kernel<<<(N0V * 64 + 255) / 256, 256, 0, stream>>>(cur0, hF0, N0V * 64);
        f32_to_f16_kernel<<<(N1V * 64 + 255) / 256, 256, 0, stream>>>(cur1, hF1, N1V * 64);

        SakeArgs a0 = {cur0, hF0, x0, N0V,
                       Wcfin[l], PF(B + 1), PF(B + 2), PF(B + 3),
                       Wo1[l], PF(B + 5), Wo2[l], PF(B + 7),
                       PF(B + 8), PF(B + 9), Wxm[l],
                       PF(B + 11), PF(B + 12), PF(B + 13), PF(B + 14),
                       PF(B + 15), PF(B + 16), PF(B + 17), PF(B + 18),
                       nxt0};
        SakeArgs a1 = a0;
        a1.h = cur1; a1.hF16 = hF1; a1.x = x1; a1.n = N1V; a1.hOut = nxt1;
        sake_layer_kernel<<<N0V, 128, sh0, stream>>>(a0);
        sake_layer_kernel<<<N1V, 128, sh1, stream>>>(a1);

        // cross-molecule attention + combine
        dot_kernel<<<(N0V * N1V + 255) / 256, 256, 0, stream>>>(nxt0, nxt1, dotb, N0V, N1V);
        msg0_kernel<<<N0V, 128, 0, stream>>>(nxt0, nxt1, dotb,
                                             PF(P_COMBINE_W), PF(P_COMBINE_B), cur0, N0V, N1V);
        msg1_kernel<<<N1V, 128, 0, stream>>>(nxt0, nxt1, dotb,
                                             PF(P_COMBINE_W), PF(P_COMBINE_B), cur1, N0V, N1V);
        // cur0/cur1 now hold the layer output; nxt* are free scratch again.
    }

    // ---------------- output head ----------------
    dense_kernel<<<(N0V * 64 + 255) / 256, 256, 0, stream>>>(
        cur0, PF(P_EMB_O1_W), PF(P_EMB_O1_B), tmp0, N0V, 64, 64, 1);
    dense_kernel<<<(N0V * 64 + 255) / 256, 256, 0, stream>>>(
        tmp0, PF(P_EMB_O2_W), PF(P_EMB_O2_B), nxt0, N0V, 64, 64, 0);
    dense_kernel<<<(N1V * 64 + 255) / 256, 256, 0, stream>>>(
        cur1, PF(P_EMB_O1_W), PF(P_EMB_O1_B), tmp1, N1V, 64, 64, 1);
    dense_kernel<<<(N1V * 64 + 255) / 256, 256, 0, stream>>>(
        tmp1, PF(P_EMB_O2_W), PF(P_EMB_O2_B), nxt1, N1V, 64, 64, 0);

    // ---------------- rigid transform ----------------
    transform_prep_kernel<<<(N0V + 63) / 64, 64, 0, stream>>>(
        nxt0, x0, PF(P_ORIG_W), PF(P_XAX_W), PF(P_YAX_W), t0, Pm0, N0V);
    transform_prep_kernel<<<(N1V + 63) / 64, 64, 0, stream>>>(
        nxt1, x1, PF(P_ORIG_W), PF(P_XAX_W), PF(P_YAX_W), t1, Pm1, N1V);

    float* out0 = (float*)d_out;
    float* out1 = out0 + (size_t)N0V * N0V * 3;
    transform_write_kernel<<<(N0V * N0V + 255) / 256, 256, 0, stream>>>(t0, Pm0, out0, N0V);
    transform_write_kernel<<<(N1V * N1V + 255) / 256, 256, 0, stream>>>(t1, Pm1, out1, N1V);
}
